// BrainGNN_7438883356948
// MI455X (gfx1250) — compile-verified, matrix operand order checked by reading, not run on hardware
//
#include <hip/hip_runtime.h>
#include <hip/hip_bf16.h>

typedef __attribute__((ext_vector_type(16))) __bf16 v16bf;
typedef __attribute__((ext_vector_type(8)))  float  v8f;

// ---------------------------------------------------------------------------
// degree / normalization
// ---------------------------------------------------------------------------
__global__ void zero_kernel(float* __restrict__ p, int n) {
    int i = blockIdx.x * blockDim.x + threadIdx.x;
    if (i < n) p[i] = 0.f;
}

__global__ void degree_kernel(const long long* __restrict__ dst, int E,
                              float* __restrict__ deg) {
    int e = blockIdx.x * blockDim.x + threadIdx.x;
    if (e < E) atomicAdd(&deg[(int)dst[e]], 1.0f);
}

__global__ void dinv_kernel(float* __restrict__ deg, int n) {
    int i = blockIdx.x * blockDim.x + threadIdx.x;
    if (i < n) deg[i] = rsqrtf(deg[i] + 1.0f);  // +1 = self loop, so deg >= 1 always
}

// ---------------------------------------------------------------------------
// Pack W[K x M] (f32, row-major) into bf16 WMMA-B lane layout:
//   Wp[((tn*ksteps + ks)*32 + lane)*16 + e] = W[k][col]
//   k   = ks*32 + (lane>=16 ? 8 : 0) + (e&7) + ((e&8) ? 16 : 0)
//   col = tn*16 + (lane&15)
// Each GEMM lane then reads its 16-element fragment as 32 contiguous bytes.
// ---------------------------------------------------------------------------
__global__ void pack_weights_bf16(const float* __restrict__ W, __bf16* __restrict__ Wp,
                                  int K, int M) {
    const int ksteps = K >> 5;
    const int tiles_n = M >> 4;
    size_t total = (size_t)tiles_n * ksteps * 32 * 16;
    size_t idx = (size_t)blockIdx.x * blockDim.x + threadIdx.x;
    if (idx >= total) return;
    int e    = (int)(idx & 15);
    int lane = (int)((idx >> 4) & 31);
    int t    = (int)(idx >> 9);          // tn*ksteps + ks
    int ks   = t % ksteps;
    int tn   = t / ksteps;
    int k    = ks * 32 + ((lane >= 16) ? 8 : 0) + (e & 7) + ((e & 8) ? 16 : 0);
    int col  = tn * 16 + (lane & 15);
    Wp[idx] = (__bf16)W[(size_t)k * M + col];
}

// ---------------------------------------------------------------------------
// C[nrows x M] = A[nrows x K] * Bp (pre-packed bf16), M = MT*16, K = KK.
// One wave computes a full 16-row x M strip. Per k-step:
//   clause of MT B-fragment loads (const-offset b128s) + 1 A load,
//   then MT back-to-back v_wmma (XDL pipelining, one wait per step).
// ---------------------------------------------------------------------------
template <int MT, int KK>
__global__ void gemm_wmma_bf16(const float* __restrict__ A, const __bf16* __restrict__ Bp,
                               const float* __restrict__ bias, float* __restrict__ C,
                               int nrows, int fuse_tanh) {
    constexpr int M      = MT * 16;
    constexpr int ksteps = KK >> 5;
    const int lane   = threadIdx.x & 31;
    const int wave   = blockIdx.x * (blockDim.x >> 5) + (threadIdx.x >> 5);
    const int tiles_m = (nrows + 15) >> 4;
    if (wave >= tiles_m) return;

    const bool hi  = lane >= 16;
    const int  l16 = lane & 15;
    const int  row = wave * 16 + l16;
    const bool rowOK = row < nrows;
    const float* __restrict__ arow = A + (size_t)row * KK;

    v8f acc[MT];
#pragma unroll
    for (int t = 0; t < MT; ++t) acc[t] = (v8f){};

#pragma unroll
    for (int ks = 0; ks < ksteps; ++ks) {
        // ISA 16-bit A layout: lanes 0-15 hold K = kb..kb+7 and kb+16..kb+23,
        // lanes 16-31 shifted by 8.
        const int kb = ks * 32 + (hi ? 8 : 0);
        float av[16];
        if (rowOK) {
            const float* p0 = arow + kb;        // 8 contiguous floats (2x b128)
            const float* p1 = arow + kb + 16;   // 8 contiguous floats (2x b128)
#pragma unroll
            for (int e = 0; e < 8; ++e) { av[e] = p0[e]; av[8 + e] = p1[e]; }
        } else {
#pragma unroll
            for (int e = 0; e < 16; ++e) av[e] = 0.f;
        }
        v16bf a;
#pragma unroll
        for (int e = 0; e < 16; ++e) a[e] = (__bf16)av[e];

        // One per-lane base pointer; per-t displacement is a compile-time
        // immediate (t * ksteps * 1024 bytes) -> clause of b128 loads.
        const v16bf* __restrict__ bbase = (const v16bf*)Bp + (size_t)ks * 32 + lane;
        v16bf bfrag[MT];
#pragma unroll
        for (int t = 0; t < MT; ++t) bfrag[t] = bbase[(size_t)t * ksteps * 32];

#pragma unroll
        for (int t = 0; t < MT; ++t)
            acc[t] = __builtin_amdgcn_wmma_f32_16x16x32_bf16(
                false, a, false, bfrag[t], (short)0, acc[t], false, false);
    }

    // D layout: lane holds column t*16 + l16; VGPR r = row wave*16 + r (+8 hi)
#pragma unroll
    for (int t = 0; t < MT; ++t) {
        const int col = t * 16 + l16;
        const float bv = bias ? bias[col] : 0.f;
#pragma unroll
        for (int r = 0; r < 8; ++r) {
            int m = wave * 16 + r + (hi ? 8 : 0);
            if (m < nrows) {
                float v = acc[t][r] + bv;
                if (fuse_tanh) v = tanhf(v);
                C[(size_t)m * M + col] = v;
            }
        }
    }
}

// ---------------------------------------------------------------------------
// agg[i,f] = h[i,f] * dinv[i]^2 + bias[f]   (self-loop term + bias init)
// ---------------------------------------------------------------------------
__global__ void self_bias_init(const float* __restrict__ h, const float* __restrict__ dinv,
                               const float* __restrict__ bias, float* __restrict__ agg,
                               int n, int F) {
    size_t idx = (size_t)blockIdx.x * blockDim.x + threadIdx.x;
    if (idx < (size_t)n * F) {
        int i = (int)(idx / F);
        int f = (int)(idx - (size_t)i * F);
        float di = dinv[i];
        agg[idx] = h[idx] * di * di + bias[f];
    }
}

// ---------------------------------------------------------------------------
// One wave per edge: lane l covers features 4l..4l+3 (full 128-float row =
// 32 lanes x float4, perfectly coalesced). 4 global_atomic_add_f32 per lane.
// Source rows and destination rows are L2-resident (25.6 MB << 192 MB L2).
// ---------------------------------------------------------------------------
__global__ void edge_scatter(const long long* __restrict__ src, const long long* __restrict__ dst,
                             const float* __restrict__ h, const float* __restrict__ dinv,
                             float* __restrict__ agg, int E) {
    int gwave = (blockIdx.x * blockDim.x + threadIdx.x) >> 5;
    int lane  = threadIdx.x & 31;
    if (gwave >= E) return;
    int s = (int)src[gwave];
    int d = (int)dst[gwave];
    float nrm = dinv[s] * dinv[d];
    float4 v = ((const float4*)(h + (size_t)s * 128))[lane];
    float* ap = agg + (size_t)d * 128 + lane * 4;
    atomicAdd(ap + 0, v.x * nrm);
    atomicAdd(ap + 1, v.y * nrm);
    atomicAdd(ap + 2, v.z * nrm);
    atomicAdd(ap + 3, v.w * nrm);
}

__global__ void tanh_kernel(float* __restrict__ p, size_t n) {
    size_t i = (size_t)blockIdx.x * blockDim.x + threadIdx.x;
    if (i < n) p[i] = tanhf(p[i]);
}

// ---------------------------------------------------------------------------
// FC2 head: one wave per node, K=64 -> 2 elements/lane + wave reduction.
// ---------------------------------------------------------------------------
__global__ void fc2_kernel(const float* __restrict__ h, const float* __restrict__ W,
                           const float* __restrict__ b, float* __restrict__ out,
                           int n, int K) {
    int gwave = (blockIdx.x * blockDim.x + threadIdx.x) >> 5;
    int lane  = threadIdx.x & 31;
    if (gwave >= n) return;
    const float* hr = h + (size_t)gwave * K;
    float sum = 0.f;
    for (int k = lane; k < K; k += 32) sum += hr[k] * W[k];
#pragma unroll
    for (int off = 16; off > 0; off >>= 1) sum += __shfl_xor(sum, off, 32);
    if (lane == 0) out[gwave] = sum + b[0];
}

// ---------------------------------------------------------------------------
extern "C" void kernel_launch(void* const* d_in, const int* in_sizes, int n_in,
                              void* d_out, int out_size, void* d_ws, size_t ws_size,
                              hipStream_t stream) {
    const float*     x    = (const float*)d_in[0];
    const long long* ei   = (const long long*)d_in[1];
    const float*     W1   = (const float*)d_in[2];
    const float*     b1   = (const float*)d_in[3];
    const float*     W2   = (const float*)d_in[4];
    const float*     b2   = (const float*)d_in[5];
    const float*     Wf1  = (const float*)d_in[6];
    const float*     bf1  = (const float*)d_in[7];
    const float*     Wf2  = (const float*)d_in[8];
    const float*     bf2  = (const float*)d_in[9];
    float* out = (float*)d_out;

    const int F  = 128;   // d_in = d1 = d2
    const int F3 = 64;    // d3
    const int N  = in_sizes[0] / F;
    const int E  = in_sizes[1] / 2;
    const long long* srcp = ei;
    const long long* dstp = ei + E;

    // workspace: dinv[N] | hA[N*128] f32 | hB[N*128] f32 | W1p | W2p | Wf1p (bf16)
    float* dinv = (float*)d_ws;
    float* hA   = dinv + (size_t)((N + 7) & ~7);            // keep 32B alignment
    float* hB   = hA + (size_t)N * F;
    __bf16* W1p  = (__bf16*)(hB + (size_t)N * F);
    const size_t wpack  = (size_t)(F / 16) * (F / 32) * 32 * 16;   // 16384 elems
    const size_t wpackf = (size_t)(F3 / 16) * (F / 32) * 32 * 16;  //  8192 elems
    __bf16* W2p  = W1p + wpack;
    __bf16* Wf1p = W2p + wpack;

    const int TB = 256;
    auto blks = [&](size_t work) { return (int)((work + TB - 1) / TB); };

    // degree -> dinv (in place)
    zero_kernel  <<<blks(N), TB, 0, stream>>>(dinv, N);
    degree_kernel<<<blks(E), TB, 0, stream>>>(dstp, E, dinv);
    dinv_kernel  <<<blks(N), TB, 0, stream>>>(dinv, N);

    // pre-pack weights into WMMA-B bf16 layout (tiny, stays hot in cache)
    pack_weights_bf16<<<blks(wpack),  TB, 0, stream>>>(W1,  W1p,  F, F);
    pack_weights_bf16<<<blks(wpack),  TB, 0, stream>>>(W2,  W2p,  F, F);
    pack_weights_bf16<<<blks(wpackf), TB, 0, stream>>>(Wf1, Wf1p, F, F3);

    const int tiles_m   = (N + 15) / 16;
    const int gemm_blks = (tiles_m + (TB / 32) - 1) / (TB / 32);
    const size_t totalF = (size_t)N * F;
    const int eg_blocks = blks((size_t)E * 32);

    // ---- GCN layer 1: hB = tanh(norm-agg(x @ W1) + b1)
    gemm_wmma_bf16<8, 128><<<gemm_blks, TB, 0, stream>>>(x, W1p, nullptr, hA, N, 0);
    self_bias_init<<<blks(totalF), TB, 0, stream>>>(hA, dinv, b1, hB, N, F);
    edge_scatter  <<<eg_blocks,    TB, 0, stream>>>(srcp, dstp, hA, dinv, hB, E);
    tanh_kernel   <<<blks(totalF), TB, 0, stream>>>(hB, totalF);

    // ---- GCN layer 2: hB = tanh(norm-agg(hB @ W2) + b2)   (hB free after gemm)
    gemm_wmma_bf16<8, 128><<<gemm_blks, TB, 0, stream>>>(hB, W2p, nullptr, hA, N, 0);
    self_bias_init<<<blks(totalF), TB, 0, stream>>>(hA, dinv, b2, hB, N, F);
    edge_scatter  <<<eg_blocks,    TB, 0, stream>>>(srcp, dstp, hA, dinv, hB, E);
    tanh_kernel   <<<blks(totalF), TB, 0, stream>>>(hB, totalF);

    // ---- FC1: hA[N x 64] = tanh(hB @ Wf1 + bf1), fused epilogue
    gemm_wmma_bf16<4, 128><<<gemm_blks, TB, 0, stream>>>(hB, Wf1p, bf1, hA, N, 1);

    // ---- FC2 head: out[N] = hA @ Wf2 + bf2
    fc2_kernel<<<blks((size_t)N * 32), TB, 0, stream>>>(hA, Wf2, bf2, out, N, F3);
}
